// CharLSTM_15934328668255
// MI455X (gfx1250) — matrix operand order repaired and checked by reading, not compiled, not run on hardware
//
#include <hip/hip_runtime.h>
#include <hip/hip_bf16.h>
#include <stddef.h>

typedef __attribute__((ext_vector_type(16))) _Float16 v16h;
typedef __attribute__((ext_vector_type(8)))  _Float16 v8h;
typedef __attribute__((ext_vector_type(8)))  float    v8f;

#define BB   256
#define TT   511
#define HHH  256
#define G4H  1024
#define VV   128

union AV { v16h v; v8h h[2]; };

// ---------------------------------------------------------------------------
// Grid-wide sense barrier (32 resident blocks; persistent kernel)
// ---------------------------------------------------------------------------
__device__ __forceinline__ void grid_barrier(unsigned* cnt, unsigned* gen, unsigned nblk) {
    __threadfence();              // publish this thread's global stores (agent scope)
    __syncthreads();
    if (threadIdx.x == 0) {
        unsigned g = __hip_atomic_load(gen, __ATOMIC_RELAXED, __HIP_MEMORY_SCOPE_AGENT);
        unsigned a = __hip_atomic_fetch_add(cnt, 1u, __ATOMIC_ACQ_REL, __HIP_MEMORY_SCOPE_AGENT);
        if (a == nblk - 1u) {
            __hip_atomic_store(cnt, 0u, __ATOMIC_RELAXED, __HIP_MEMORY_SCOPE_AGENT);
            __hip_atomic_store(gen, g + 1u, __ATOMIC_RELEASE, __HIP_MEMORY_SCOPE_AGENT);
        } else {
            while (__hip_atomic_load(gen, __ATOMIC_ACQUIRE, __HIP_MEMORY_SCOPE_AGENT) == g) {
                __builtin_amdgcn_s_sleep(2);
            }
        }
    }
    __syncthreads();
    __threadfence();              // acquire others' stores
}

__device__ __forceinline__ float sigmoidf_(float x) {
    return 1.0f / (1.0f + __expf(-x));
}

// ---------------------------------------------------------------------------
// Precompute: ctxW[b, n] = bias[n] + ctx48(b) . W[0:48, n]
// ---------------------------------------------------------------------------
__global__ void ctxw_kernel(const int* __restrict__ rhythm, const int* __restrict__ meter,
                            const int* __restrict__ key,
                            const float* __restrict__ er, const float* __restrict__ em,
                            const float* __restrict__ ek, const float* __restrict__ W,
                            const float* __restrict__ bias, float* __restrict__ ctxW) {
    int idx = blockIdx.x * blockDim.x + threadIdx.x;
    if (idx >= BB * G4H) return;
    int b = idx >> 10, n = idx & (G4H - 1);
    const float* r = er + rhythm[b] * 16;
    const float* m = em + meter[b]  * 16;
    const float* k = ek + key[b]    * 16;
    float s = bias[n];
    #pragma unroll
    for (int d = 0; d < 16; ++d) s += r[d] * W[d * G4H + n];
    #pragma unroll
    for (int d = 0; d < 16; ++d) s += m[d] * W[(16 + d) * G4H + n];
    #pragma unroll
    for (int d = 0; d < 16; ++d) s += k[d] * W[(32 + d) * G4H + n];
    ctxW[idx] = s;
}

// ---------------------------------------------------------------------------
// Precompute: embW[tok, n] = emb_tune[tok, :] . W[48:80, n]
// ---------------------------------------------------------------------------
__global__ void embw_kernel(const float* __restrict__ et, const float* __restrict__ W,
                            float* __restrict__ embW) {
    int idx = blockIdx.x * blockDim.x + threadIdx.x;
    if (idx >= 129 * G4H) return;
    int tok = idx >> 10, n = idx & (G4H - 1);
    float s = 0.f;
    #pragma unroll
    for (int d = 0; d < 32; ++d) s += et[tok * 32 + d] * W[(48 + d) * G4H + n];
    embW[idx] = s;
}

// ---------------------------------------------------------------------------
// Swizzle a row-major fp32 matrix (8*32 rows x NT*16 cols) into the WMMA
// B-operand lane layout, as f16. Lanes 0-15: K 0..7,16..23, lanes 16-31: K 8..15,24..31.
// out element index == ((kc*NT + nt)*32 + lane)*16 + e
// ---------------------------------------------------------------------------
__global__ void swz_kernel(const float* __restrict__ M, _Float16* __restrict__ out,
                           int NT, int ldn) {
    int idx = blockIdx.x * blockDim.x + threadIdx.x;
    int total = 8 * NT * 512;
    if (idx >= total) return;
    int e    = idx & 15;
    int lane = (idx >> 4) & 31;
    int rest = idx >> 9;
    int nt   = rest % NT;
    int kc   = rest / NT;
    int k = kc * 32 + (lane >> 4) * 8 + (e & 7) + (e >> 3) * 16;
    int n = nt * 16 + (lane & 15);
    out[idx] = (_Float16)M[k * ldn + n];
}

__global__ void init_kernel(unsigned* bar) {
    if (threadIdx.x < 2) bar[threadIdx.x] = 0u;
}

// ---------------------------------------------------------------------------
// Persistent LSTM recurrence. 32 blocks x 256 threads = 256 waves.
// Wave w: bt = w&15 (batch tile), ht = w>>4 (h-column tile, uniform per block).
// c/h state lives in registers; h exchanged via double-buffered global f16.
// ---------------------------------------------------------------------------
__global__ __launch_bounds__(256, 1)
void lstm_persistent(const int* __restrict__ tune, const int* __restrict__ tune_len,
                     const float* __restrict__ ctxW, const float* __restrict__ embW,
                     const _Float16* __restrict__ Uswz, _Float16* __restrict__ hbuf,
                     _Float16* __restrict__ ys, unsigned* __restrict__ bar) {
    __shared__ _Float16 ldsU[4 * 8 * 32 * 16];   // 32 KB: U tiles for this block's ht

    const int lane = threadIdx.x & 31;
    const int wave = threadIdx.x >> 5;
    const int wgid = blockIdx.x * 8 + wave;      // 0..255
    const int bt   = wgid & 15;
    const int ht   = blockIdx.x >> 1;            // uniform per block (== wgid>>4)
    const int hi   = lane >> 4;                  // 0/1: lane half
    const int n0   = lane & 15;

    // ---- stage U tiles (gate g, k-chunk kc) for this ht into LDS ----
    {
        const v8h* src = (const v8h*)Uswz;
        v8h* dst = (v8h*)ldsU;
        for (int j = threadIdx.x; j < 2048; j += 256) {
            int tileL = j >> 6;                  // g*8 + kc
            int g  = tileL >> 3, kc = tileL & 7;
            int inner = j & 63;
            dst[j] = src[(size_t)(kc * 64 + g * 16 + ht) * 64 + inner];
        }
    }
    __syncthreads();

    // ---- zero h buffer 0 (this wave's 16x16 tile) ----
    {
        v8h z = {};
        *(v8h*)&hbuf[(size_t)(bt * 16 + n0) * HHH + ht * 16 + hi * 8] = z;
    }

    // ---- per-wave persistent state ----
    v8f c_state = {0.f, 0.f, 0.f, 0.f, 0.f, 0.f, 0.f, 0.f};
    v8f h_state = {0.f, 0.f, 0.f, 0.f, 0.f, 0.f, 0.f, 0.f};

    float ctx0[4][8];
    int   lens[8];
    int   brows[8];
    #pragma unroll
    for (int v = 0; v < 8; ++v) {
        int brow = bt * 16 + v + hi * 8;         // C-tile row M = v + 8*hi
        brows[v] = brow;
        lens[v]  = tune_len[brow];
        #pragma unroll
        for (int g = 0; g < 4; ++g)
            ctx0[g][v] = ctxW[(size_t)brow * G4H + g * 256 + ht * 16 + n0];
    }

    unsigned* cnt = bar;
    unsigned* gen = bar + 1;
    grid_barrier(cnt, gen, gridDim.x);           // h0 visible everywhere

    for (int t = 0; t < TT; ++t) {
        const _Float16* hr = hbuf + (size_t)(t & 1) * (BB * HHH);
        _Float16*       hw = hbuf + (size_t)((t + 1) & 1) * (BB * HHH);

        // z init = ctxW + embW[token]  (per-lane fp32 gathers)
        v8f acc0, acc1, acc2, acc3;
        #pragma unroll
        for (int v = 0; v < 8; ++v) {
            int tok = tune[brows[v] * TT + t];
            const float* eb = embW + (size_t)tok * G4H + ht * 16 + n0;
            acc0[v] = ctx0[0][v] + eb[0];
            acc1[v] = ctx0[1][v] + eb[256];
            acc2[v] = ctx0[2][v] + eb[512];
            acc3[v] = ctx0[3][v] + eb[768];
        }

        // z += h @ U   (A from row-major f16 h, B from LDS)
        const _Float16* arow = hr + (size_t)(bt * 16 + n0) * HHH + hi * 8;
        #pragma unroll
        for (int kc = 0; kc < 8; ++kc) {
            AV a;
            a.h[0] = *(const v8h*)(arow + kc * 32);        // K 0..7  (+hi*8)
            a.h[1] = *(const v8h*)(arow + kc * 32 + 16);   // K 16..23(+hi*8)
            const v16h* b0 = (const v16h*)&ldsU[((0 * 8 + kc) * 32 + lane) * 16];
            const v16h* b1 = (const v16h*)&ldsU[((1 * 8 + kc) * 32 + lane) * 16];
            const v16h* b2 = (const v16h*)&ldsU[((2 * 8 + kc) * 32 + lane) * 16];
            const v16h* b3 = (const v16h*)&ldsU[((3 * 8 + kc) * 32 + lane) * 16];
            acc0 = __builtin_amdgcn_wmma_f32_16x16x32_f16(false, a.v, false, *b0, (short)0, acc0, false, false);
            acc1 = __builtin_amdgcn_wmma_f32_16x16x32_f16(false, a.v, false, *b1, (short)0, acc1, false, false);
            acc2 = __builtin_amdgcn_wmma_f32_16x16x32_f16(false, a.v, false, *b2, (short)0, acc2, false, false);
            acc3 = __builtin_amdgcn_wmma_f32_16x16x32_f16(false, a.v, false, *b3, (short)0, acc3, false, false);
        }

        // gates, state update, masked outputs
        #pragma unroll
        for (int v = 0; v < 8; ++v) {
            float iv = sigmoidf_(acc0[v]);
            float fv = sigmoidf_(acc1[v]);
            float gv = tanhf(acc2[v]);
            float ov = sigmoidf_(acc3[v]);
            float cn = fv * c_state[v] + iv * gv;
            float hn = ov * tanhf(cn);
            bool  m  = t < lens[v];
            float h2 = m ? hn : h_state[v];
            c_state[v] = m ? cn : c_state[v];
            h_state[v] = h2;
            int brow = brows[v];
            hw[(size_t)brow * HHH + ht * 16 + n0] = (_Float16)h2;
            ys[((size_t)t * BB + brow) * HHH + ht * 16 + n0] = (_Float16)(m ? hn : 0.f);
        }

        grid_barrier(cnt, gen, gridDim.x);
    }
}

// ---------------------------------------------------------------------------
// Projection: out[b,t,:] = ys[t,b,:] @ Wd + bd. One wave per 16x16 C tile.
// ---------------------------------------------------------------------------
__global__ __launch_bounds__(256)
void proj_kernel(const _Float16* __restrict__ ys, const _Float16* __restrict__ Wdswz,
                 const float* __restrict__ bd, float* __restrict__ out) {
    const int lane = threadIdx.x & 31;
    const int wid  = (int)((blockIdx.x * blockDim.x + threadIdx.x) >> 5);   // wave-uniform
    if (wid >= TT * 16 * 8) return;                                         // wave-uniform exit
    const int nt    = wid & 7;
    const int btile = (wid >> 3) & 15;
    const int t     = wid >> 7;
    const int hi = lane >> 4, n0 = lane & 15;

    float bdv = bd[nt * 16 + n0];
    v8f acc = {bdv, bdv, bdv, bdv, bdv, bdv, bdv, bdv};

    const _Float16* arow = ys + ((size_t)t * BB + btile * 16 + n0) * HHH + hi * 8;
    #pragma unroll
    for (int kc = 0; kc < 8; ++kc) {
        AV a;
        a.h[0] = *(const v8h*)(arow + kc * 32);
        a.h[1] = *(const v8h*)(arow + kc * 32 + 16);
        const v16h* bp = (const v16h*)&Wdswz[((size_t)(kc * 8 + nt) * 32 + lane) * 16];
        acc = __builtin_amdgcn_wmma_f32_16x16x32_f16(false, a.v, false, *bp, (short)0, acc, false, false);
    }
    #pragma unroll
    for (int v = 0; v < 8; ++v) {
        int b = btile * 16 + v + hi * 8;
        out[((size_t)b * TT + t) * VV + nt * 16 + n0] = acc[v];
    }
}

// ---------------------------------------------------------------------------
extern "C" void kernel_launch(void* const* d_in, const int* in_sizes, int n_in,
                              void* d_out, int out_size, void* d_ws, size_t ws_size,
                              hipStream_t stream) {
    (void)in_sizes; (void)n_in; (void)out_size; (void)ws_size;
    const int*   tune     = (const int*)  d_in[0];
    const int*   rhythm   = (const int*)  d_in[1];
    const int*   meter    = (const int*)  d_in[2];
    const int*   key      = (const int*)  d_in[3];
    const int*   tune_len = (const int*)  d_in[4];
    const float* er       = (const float*)d_in[5];
    const float* em       = (const float*)d_in[6];
    const float* ek       = (const float*)d_in[7];
    const float* et       = (const float*)d_in[8];
    const float* W        = (const float*)d_in[9];
    const float* U        = (const float*)d_in[10];
    const float* bias     = (const float*)d_in[11];
    const float* Wd       = (const float*)d_in[12];
    const float* bd       = (const float*)d_in[13];

    // ---- workspace layout (all offsets 256B aligned), ~69.4 MB total ----
    char* ws = (char*)d_ws;
    unsigned* bar    = (unsigned*)ws;                       //   256 B
    float*    ctxW   = (float*)   (ws + 256);               //  1,048,576 B
    float*    embW   = (float*)   (ws + 256 + 1048576);     //    528,384 B
    _Float16* Uswz   = (_Float16*)(ws + 256 + 1048576 + 528384);          // 524,288 B
    _Float16* Wdswz  = (_Float16*)((char*)Uswz  + 524288);  //     65,536 B
    _Float16* hbuf   = (_Float16*)((char*)Wdswz + 65536);   //    262,144 B (2x h)
    _Float16* ysbuf  = (_Float16*)((char*)hbuf  + 262144);  // 66,977,792 B

    init_kernel<<<1, 64, 0, stream>>>(bar);
    ctxw_kernel<<<(BB * G4H + 255) / 256, 256, 0, stream>>>(rhythm, meter, key,
                                                            er, em, ek, W, bias, ctxW);
    embw_kernel<<<(129 * G4H + 255) / 256, 256, 0, stream>>>(et, W, embW);
    swz_kernel<<<(8 * 64 * 512 + 255) / 256, 256, 0, stream>>>(U,  Uswz, 64, G4H);
    swz_kernel<<<(8 * 8  * 512 + 255) / 256, 256, 0, stream>>>(Wd, Wdswz, 8, VV);

    lstm_persistent<<<32, 256, 0, stream>>>(tune, tune_len, ctxW, embW,
                                            Uswz, hbuf, ysbuf, bar);

    proj_kernel<<<(TT * 16 * 8) / 8, 256, 0, stream>>>(ysbuf, Wdswz, bd, (float*)d_out);
}
